// PCmer_51685636440485
// MI455X (gfx1250) — matrix operand order repaired and checked by reading, not compiled, not run on hardware
//
#include <hip/hip_runtime.h>
#include <hip/hip_bf16.h>
#include <math.h>

typedef __attribute__((ext_vector_type(16))) _Float16 v16h;
typedef __attribute__((ext_vector_type(8)))  _Float16 h8;
typedef __attribute__((ext_vector_type(8)))  float    v8f;

#define LL 6
#define BB 4
#define NN 2048
#define DDIM 512
#define HH 8
#define DHH 64
#define IDD 512
#define MMF 266
#define MP  384          // MMF padded to a multiple of 128 -> guard-free GEMMs
#define INNER_C 1024
#define KW 31

// ---------------------------------------------------------------------------
// WMMA helper
// ---------------------------------------------------------------------------
__device__ __forceinline__ v8f wmma_f16(v16h a, v16h b, v8f c) {
  return __builtin_amdgcn_wmma_f32_16x16x32_f16(
      false, a, false, b, (short)0, c, false, false);
}

// A fragment (16x32 f16). AT=false: A row-major Mrows x K (contiguous b128
// loads). AT=true: A stored K x Mrows (strided 16-bit loads). No guards.
template<bool AT>
__device__ __forceinline__ v16h load_a16(const _Float16* __restrict__ A,
                                         int lda, int mBase, int kBase, int lane) {
  v16h a;
  int lr = lane & 15, hf = lane >> 4;
  if (!AT) {
    const _Float16* p = A + (size_t)(mBase + lr) * lda + kBase + hf * 8;
    h8 lo = *(const h8*)p;
    h8 hi = *(const h8*)(p + 16);
#pragma unroll
    for (int j = 0; j < 8; ++j) { a[j] = lo[j]; a[8 + j] = hi[j]; }
  } else {
    int row = mBase + lr;
#pragma unroll
    for (int j = 0; j < 16; ++j) {
      int koff = (j < 8) ? (hf ? (8 + j) : j) : (hf ? (16 + j) : (8 + j));
      a[j] = A[(size_t)(kBase + koff) * lda + row];
    }
  }
  return a;
}

// B fragment (32x16 f16). BT=true: weight W[N x K] row-major (contiguous b128
// loads of W^T). BT=false: B stored K x N (strided 16-bit loads). No guards.
template<bool BT>
__device__ __forceinline__ v16h load_b16(const _Float16* __restrict__ B,
                                         int ldb, int nBase, int kBase, int lane) {
  v16h b;
  int lr = lane & 15, hf = lane >> 4;
  int col = nBase + lr;
  if (BT) {
    const _Float16* p = B + (size_t)col * ldb + kBase + hf * 8;
    h8 lo = *(const h8*)p;
    h8 hi = *(const h8*)(p + 16);
#pragma unroll
    for (int j = 0; j < 8; ++j) { b[j] = lo[j]; b[8 + j] = hi[j]; }
  } else {
#pragma unroll
    for (int j = 0; j < 16; ++j)
      b[j] = B[(size_t)(kBase + hf * 8 + (j < 8 ? j : 8 + j)) * ldb + col];
  }
  return b;
}

// ---------------------------------------------------------------------------
// Guard-free batched GEMM. Block = 128 threads = 4 waves (2x2), block tile
// 128x64. Each wave owns a 64x32 tile (4x2 WMMA fragments, f32 accumulation):
// 8 WMMAs per 6KB of fragment traffic, B-fragments reused 4x.
//   val = alpha*acc * rowscale[row] + bias[col] + resid[row,col]
// Batch z -> (b = z/batchH, h = z%batchH), per-operand (sb,sh) strides.
// All launch shapes are exact multiples of the tile; K multiple of 32.
// ---------------------------------------------------------------------------
template<bool AT, bool BT>
__global__ void __launch_bounds__(128)
gemm_wmma(const _Float16* __restrict__ A, int lda, long long sAb, long long sAh,
          const _Float16* __restrict__ Bm, int ldb, long long sBb, long long sBh,
          const float* __restrict__ bias,
          const float* __restrict__ resid, long long sRb, long long sRh,
          const float* __restrict__ rowscale, long long sSb, long long sSh,
          float alpha,
          float* __restrict__ C, _Float16* __restrict__ C16,
          int ldc, long long sCb, long long sCh,
          int Kdim, int batchH) {
  int z = blockIdx.z;
  int bb = z / batchH, hh = z % batchH;
  const _Float16* Ab = A + (long long)bb * sAb + (long long)hh * sAh;
  const _Float16* Bb = Bm + (long long)bb * sBb + (long long)hh * sBh;
  float*     Cb  = C   ? C   + (long long)bb * sCb + (long long)hh * sCh : nullptr;
  _Float16*  Cb16= C16 ? C16 + (long long)bb * sCb + (long long)hh * sCh : nullptr;
  const float* Rb = resid ? resid + (long long)bb * sRb + (long long)hh * sRh : nullptr;
  const float* Sb = rowscale ? rowscale + (long long)bb * sSb + (long long)hh * sSh : nullptr;

  int wave = threadIdx.x >> 5;
  int lane = threadIdx.x & 31;
  int wm = wave >> 1, wn = wave & 1;
  int mBase = blockIdx.y * 128 + wm * 64;
  int nBase = blockIdx.x * 64 + wn * 32;

  v8f acc[4][2] = {};
  for (int k0 = 0; k0 < Kdim; k0 += 32) {
    v16h b0 = load_b16<BT>(Bb, ldb, nBase,      k0, lane);
    v16h b1 = load_b16<BT>(Bb, ldb, nBase + 16, k0, lane);
#pragma unroll
    for (int i = 0; i < 4; ++i) {
      v16h a = load_a16<AT>(Ab, lda, mBase + i * 16, k0, lane);
      acc[i][0] = wmma_f16(a, b0, acc[i][0]);
      acc[i][1] = wmma_f16(a, b1, acc[i][1]);
    }
  }

  int lr = lane & 15, hf = lane >> 4;
#pragma unroll
  for (int i = 0; i < 4; ++i) {
#pragma unroll
    for (int j = 0; j < 2; ++j) {
      int mB = mBase + i * 16;
      int nB = nBase + j * 16;
      int col = nB + lr;
#pragma unroll
      for (int r = 0; r < 8; ++r) {
        int row = mB + hf * 8 + r;
        float val = alpha * acc[i][j][r];
        if (Sb)   val *= Sb[row];
        if (bias) val += bias[col];
        if (Rb)   val += Rb[(size_t)row * ldc + col];
        if (Cb)   Cb[(size_t)row * ldc + col] = val;
        if (Cb16) Cb16[(size_t)row * ldc + col] = (_Float16)val;
      }
    }
  }
}

// ---------------------------------------------------------------------------
// LayerNorm over D=512; emits f32 + f16 copies.
// ---------------------------------------------------------------------------
__global__ void __launch_bounds__(256)
ln_kernel(const float* __restrict__ x, const float* __restrict__ g,
          const float* __restrict__ b, float* __restrict__ out,
          _Float16* __restrict__ out16) {
  __shared__ float red[256];
  int row = blockIdx.x, t = threadIdx.x;
  const float* xr = x + (size_t)row * DDIM;
  float a0 = xr[t], a1 = xr[t + 256];
  red[t] = a0 + a1;
  __syncthreads();
  for (int off = 128; off > 0; off >>= 1) {
    if (t < off) red[t] += red[t + off];
    __syncthreads();
  }
  float mu = red[0] * (1.0f / DDIM);
  __syncthreads();
  float d0 = a0 - mu, d1 = a1 - mu;
  red[t] = d0 * d0 + d1 * d1;
  __syncthreads();
  for (int off = 128; off > 0; off >>= 1) {
    if (t < off) red[t] += red[t + off];
    __syncthreads();
  }
  float inv = rsqrtf(red[0] * (1.0f / DDIM) + 1e-5f);
  float o0 = d0 * inv * g[t] + b[t];
  float o1 = d1 * inv * g[t + 256] + b[t + 256];
  out[(size_t)row * DDIM + t] = o0;
  out[(size_t)row * DDIM + t + 256] = o1;
  out16[(size_t)row * DDIM + t] = (_Float16)o0;
  out16[(size_t)row * DDIM + t + 256] = (_Float16)o1;
}

// generic f32 -> f16 convert
__global__ void cvt_f16(const float* __restrict__ src, _Float16* __restrict__ dst,
                        long long n) {
  long long i = (long long)blockIdx.x * blockDim.x + threadIdx.x;
  if (i < n) dst[i] = (_Float16)src[i];
}

// pack proj (266x64) into zero-padded f16 (MP x 64)
__global__ void pack_proj(const float* __restrict__ src, _Float16* __restrict__ dst) {
  int i = blockIdx.x * blockDim.x + threadIdx.x;
  if (i >= MP * DHH) return;
  int m = i / DHH;
  dst[i] = (m < MMF) ? (_Float16)src[i] : (_Float16)0.f;
}

// ---------------------------------------------------------------------------
// FAVOR+ feature map: dd[bh,n,0:MP] holds dn*(data @ proj^T) (pad cols = 0).
// Writes f16 feature map with zeroed pad columns.
// ---------------------------------------------------------------------------
__global__ void __launch_bounds__(128)
favor_phase2(const float* __restrict__ data, const float* __restrict__ dd,
             _Float16* __restrict__ fp, int isQuery) {
  __shared__ float red[128];
  int idx = blockIdx.x;                 // over B*H*N (n fastest)
  int n = idx % NN, bh = idx / NN;
  int h = bh % HH, b = bh / HH;
  const float* dp = data + ((size_t)b * NN + n) * IDD + h * DHH;
  int t = threadIdx.x;
  float s = 0.f;
  for (int i = t; i < DHH; i += 128) { float v = dp[i]; s += v * v; }
  red[t] = s;
  __syncthreads();
  for (int off = 64; off > 0; off >>= 1) {
    if (t < off) red[t] += red[t + off];
    __syncthreads();
  }
  float diag = red[0] * 0.0625f;        // 0.5 * dn^2 = 1/16
  __syncthreads();
  const float* row = dd + (size_t)idx * MP;
  _Float16* orow = fp + (size_t)idx * MP;
  float mx = 0.f;
  if (isQuery) {
    float m = -1e30f;
    for (int m2 = t; m2 < MMF; m2 += 128) m = fmaxf(m, row[m2]);
    red[t] = m;
    __syncthreads();
    for (int off = 64; off > 0; off >>= 1) {
      if (t < off) red[t] = fmaxf(red[t], red[t + off]);
      __syncthreads();
    }
    mx = red[0];
  }
  const float ratio = 0.06131393f;      // 266^-0.5
  for (int m2 = t; m2 < MP; m2 += 128) {
    float r = 0.f;
    if (m2 < MMF) {
      float v = row[m2];
      r = isQuery ? ratio * (__expf(v - diag - mx) + 1e-4f)
                  : ratio * __expf(v - diag + 1e-4f);
    }
    orow[m2] = (_Float16)r;
  }
}

// ksum[bh,m] = sum_n kp16[bh,n,m]
__global__ void ksum_kernel(const _Float16* __restrict__ kp, float* __restrict__ ksum) {
  int i = blockIdx.x * blockDim.x + threadIdx.x;
  if (i >= BB * HH * MMF) return;
  int bh = i / MMF, m = i % MMF;
  const _Float16* base = kp + (size_t)bh * NN * MP + m;
  float s = 0.f;
  for (int n = 0; n < NN; ++n) s += (float)base[(size_t)n * MP];
  ksum[(size_t)bh * MMF + m] = s;
}

// dinv[bh,n] = 1 / (qp16[bh,n,:].ksum[bh,:] + 1e-8); one wave32 per row
__global__ void __launch_bounds__(128)
dinv_kernel(const _Float16* __restrict__ qp, const float* __restrict__ ksum,
            float* __restrict__ dinv) {
  int w = threadIdx.x >> 5, lane = threadIdx.x & 31;
  long long row = (long long)blockIdx.x * 4 + w;
  if (row >= (long long)BB * HH * NN) return;
  int bh = (int)(row / NN);
  const _Float16* qrow = qp + (size_t)row * MP;
  const float* ks = ksum + (size_t)bh * MMF;
  float s = 0.f;
  for (int m = lane; m < MMF; m += 32) s += (float)qrow[m] * ks[m];
#pragma unroll
  for (int off = 16; off > 0; off >>= 1) s += __shfl_xor(s, off, 32);
  if (lane == 0) dinv[row] = 1.f / (s + 1e-8f);
}

// GLU
__global__ void glu_kernel(const float* __restrict__ y, float* __restrict__ y2) {
  long long i = (long long)blockIdx.x * blockDim.x + threadIdx.x;
  if (i >= (long long)BB * NN * INNER_C) return;
  long long row = i / INNER_C; int c = (int)(i % INNER_C);
  float a = y[row * (2 * INNER_C) + c];
  float g = y[row * (2 * INNER_C) + INNER_C + c];
  y2[i] = a * (1.f / (1.f + __expf(-g)));
}

// Depthwise conv (K=31, pad 15) + bias + SiLU -> f16 (GEMM operand)
__global__ void dwconv_kernel(const float* __restrict__ y2,
                              const float* __restrict__ w,
                              const float* __restrict__ bcoef,
                              _Float16* __restrict__ out16) {
  long long i = (long long)blockIdx.x * blockDim.x + threadIdx.x;
  if (i >= (long long)BB * NN * INNER_C) return;
  int c = (int)(i % INNER_C);
  int n = (int)((i / INNER_C) % NN);
  int b = (int)(i / ((long long)INNER_C * NN));
  const float* wr = w + (size_t)c * KW;
  float s = bcoef[c];
#pragma unroll
  for (int k = 0; k < KW; ++k) {
    int nn = n + k - 15;
    if (nn >= 0 && nn < NN)
      s += y2[((size_t)b * NN + nn) * INNER_C + c] * wr[k];
  }
  out16[i] = (_Float16)(s * (1.f / (1.f + __expf(-s))));
}

// ---------------------------------------------------------------------------
extern "C" void kernel_launch(void* const* d_in, const int* in_sizes, int n_in,
                              void* d_out, int out_size, void* d_ws, size_t ws_size,
                              hipStream_t stream) {
  const float* x_in  = (const float*)d_in[0];
  const float* ln1_g = (const float*)d_in[1];
  const float* ln1_b = (const float*)d_in[2];
  const float* wq    = (const float*)d_in[3];
  const float* bq    = (const float*)d_in[4];
  const float* wk    = (const float*)d_in[5];
  const float* bk    = (const float*)d_in[6];
  const float* wv    = (const float*)d_in[7];
  const float* bv    = (const float*)d_in[8];
  const float* wo    = (const float*)d_in[9];
  const float* bo    = (const float*)d_in[10];
  const float* proj  = (const float*)d_in[11];
  const float* ln2_g = (const float*)d_in[12];
  const float* ln2_b = (const float*)d_in[13];
  const float* pw1_w = (const float*)d_in[14];
  const float* pw1_b = (const float*)d_in[15];
  const float* dw_w  = (const float*)d_in[16];
  const float* dw_b  = (const float*)d_in[17];
  const float* pw2_w = (const float*)d_in[18];
  const float* pw2_b = (const float*)d_in[19];

  const size_t R    = (size_t)BB * NN;                 // 8192
  const size_t RD   = R * DDIM;                        // 4,194,304
  const size_t FPsz = (size_t)BB * HH * NN * MP;       // 25,165,824
  const int    BH   = BB * HH;

  // ---- workspace layout -------------------------------------------------
  char* base = (char*)d_ws;
  size_t off = 0;
  auto alloc_f32 = [&](size_t n) { float* p = (float*)(base + off);
                                   off += ((n * 4 + 255) & ~(size_t)255); return p; };
  auto alloc_f16 = [&](size_t n) { _Float16* p = (_Float16*)(base + off);
                                   off += ((n * 2 + 255) & ~(size_t)255); return p; };
  float* xbuf = alloc_f32(RD);
  float* hbuf = alloc_f32(RD);
  float* qb   = alloc_f32(RD);
  float* kb   = alloc_f32(RD);     // qb..kb contiguous: y2 alias (2*RD == R*INNER)
  float* dd   = alloc_f32(FPsz);   // raw feature scores; also aliased by y (pw1 out)
  float* ksum = alloc_f32((size_t)BH * MMF);
  float* dinv = alloc_f32((size_t)BH * NN);
  _Float16* h16   = alloc_f16(RD);
  _Float16* q16   = alloc_f16(RD);  // also aliased by o16 (attention output)
  _Float16* k16   = alloc_f16(RD);
  _Float16* v16b  = alloc_f16(RD);
  _Float16* qp16  = alloc_f16(FPsz);
  _Float16* kp16  = alloc_f16(FPsz);
  _Float16* ctx16 = alloc_f16((size_t)BH * MP * DHH);
  _Float16* dw16  = alloc_f16(R * INNER_C);
  _Float16* wq16  = alloc_f16((size_t)IDD * DDIM);
  _Float16* wk16  = alloc_f16((size_t)IDD * DDIM);
  _Float16* wv16  = alloc_f16((size_t)IDD * DDIM);
  _Float16* wo16  = alloc_f16((size_t)IDD * DDIM);
  _Float16* pj16  = alloc_f16((size_t)MP * DHH);
  _Float16* p116  = alloc_f16((size_t)2 * INNER_C * DDIM);
  _Float16* p216  = alloc_f16((size_t)DDIM * INNER_C);
  float* yb  = dd;                 // pw1 output (R x 2*INNER <= FPsz)
  float* y2b = qb;                 // GLU output (R x INNER == 2*RD)
  _Float16* o16 = q16;             // attention output reuses q16

  hipMemcpyAsync(xbuf, x_in, RD * sizeof(float), hipMemcpyDeviceToDevice, stream);

  const float dn = 0.35355339059f;  // 64^-0.25
  dim3 blk128(128), blk256(256);
  const long long WSQ = (long long)IDD * DDIM;

  for (int l = 0; l < LL; ++l) {
    // ---- per-layer weight conversion to f16 ----
    cvt_f16<<<dim3((unsigned)((WSQ + 255) / 256)), blk256, 0, stream>>>(
        wq + (size_t)l * WSQ, wq16, WSQ);
    cvt_f16<<<dim3((unsigned)((WSQ + 255) / 256)), blk256, 0, stream>>>(
        wk + (size_t)l * WSQ, wk16, WSQ);
    cvt_f16<<<dim3((unsigned)((WSQ + 255) / 256)), blk256, 0, stream>>>(
        wv + (size_t)l * WSQ, wv16, WSQ);
    cvt_f16<<<dim3((unsigned)((WSQ + 255) / 256)), blk256, 0, stream>>>(
        wo + (size_t)l * WSQ, wo16, WSQ);
    pack_proj<<<dim3((MP * DHH + 255) / 256), blk256, 0, stream>>>(
        proj + (size_t)l * MMF * DHH, pj16);
    {
      long long n1 = (long long)2 * INNER_C * DDIM;
      cvt_f16<<<dim3((unsigned)((n1 + 255) / 256)), blk256, 0, stream>>>(
          pw1_w + (size_t)l * n1, p116, n1);
      long long n2 = (long long)DDIM * INNER_C;
      cvt_f16<<<dim3((unsigned)((n2 + 255) / 256)), blk256, 0, stream>>>(
          pw2_w + (size_t)l * n2, p216, n2);
    }

    // ---- LN1 ----
    ln_kernel<<<dim3((unsigned)R), blk256, 0, stream>>>(
        xbuf, ln1_g + l * DDIM, ln1_b + l * DDIM, hbuf, h16);

    // ---- Q, K, V projections ----
    {
      dim3 grid(IDD / 64, (unsigned)(R / 128), 1);
      gemm_wmma<false, true><<<grid, blk128, 0, stream>>>(
          h16, DDIM, 0, 0, wq16, DDIM, 0, 0, bq + (size_t)l * IDD,
          nullptr, 0, 0, nullptr, 0, 0, 1.f, qb, q16, IDD, 0, 0, DDIM, 1);
      gemm_wmma<false, true><<<grid, blk128, 0, stream>>>(
          h16, DDIM, 0, 0, wk16, DDIM, 0, 0, bk + (size_t)l * IDD,
          nullptr, 0, 0, nullptr, 0, 0, 1.f, kb, k16, IDD, 0, 0, DDIM, 1);
      gemm_wmma<false, true><<<grid, blk128, 0, stream>>>(
          h16, DDIM, 0, 0, wv16, DDIM, 0, 0, bv + (size_t)l * IDD,
          nullptr, 0, 0, nullptr, 0, 0, 1.f, nullptr, v16b, IDD, 0, 0, DDIM, 1);
    }

    // ---- FAVOR feature maps (q then k, sharing the dd scratch) ----
    {
      dim3 grid(MP / 64, NN / 128, BH);
      gemm_wmma<false, true><<<grid, blk128, 0, stream>>>(
          q16, IDD, (long long)NN * IDD, DHH, pj16, DHH, 0, 0, nullptr,
          nullptr, 0, 0, nullptr, 0, 0, dn, dd, nullptr, MP,
          (long long)HH * NN * MP, (long long)NN * MP, DHH, HH);
      favor_phase2<<<dim3(BH * NN), blk128, 0, stream>>>(qb, dd, qp16, 1);
      gemm_wmma<false, true><<<grid, blk128, 0, stream>>>(
          k16, IDD, (long long)NN * IDD, DHH, pj16, DHH, 0, 0, nullptr,
          nullptr, 0, 0, nullptr, 0, 0, dn, dd, nullptr, MP,
          (long long)HH * NN * MP, (long long)NN * MP, DHH, HH);
      favor_phase2<<<dim3(BH * NN), blk128, 0, stream>>>(kb, dd, kp16, 0);
    }
    ksum_kernel<<<dim3((BH * MMF + 255) / 256), blk256, 0, stream>>>(kp16, ksum);

    // ---- context = kp^T @ v : [MP x NN] @ [NN x DH] ----
    {
      dim3 grid(1, MP / 128, BH);
      gemm_wmma<true, false><<<grid, blk128, 0, stream>>>(
          kp16, MP, (long long)HH * NN * MP, (long long)NN * MP,
          v16b, IDD, (long long)NN * IDD, DHH, nullptr,
          nullptr, 0, 0, nullptr, 0, 0, 1.f, nullptr, ctx16, DHH,
          (long long)HH * MP * DHH, (long long)MP * DHH, NN, HH);
    }
    dinv_kernel<<<dim3((BH * NN + 3) / 4), blk128, 0, stream>>>(qp16, ksum, dinv);

    // ---- out = dinv * (qp @ context) -> (B,N,H,DH) f16 ----
    {
      dim3 grid(1, NN / 128, BH);
      gemm_wmma<false, false><<<grid, blk128, 0, stream>>>(
          qp16, MP, (long long)HH * NN * MP, (long long)NN * MP,
          ctx16, DHH, (long long)HH * MP * DHH, (long long)MP * DHH, nullptr,
          nullptr, 0, 0, dinv, (long long)HH * NN, (long long)NN, 1.f,
          nullptr, o16, IDD, (long long)NN * IDD, DHH, MP, HH);
    }

    // ---- x = x + o @ wo^T + bo ----
    {
      dim3 grid(DDIM / 64, (unsigned)(R / 128), 1);
      gemm_wmma<false, true><<<grid, blk128, 0, stream>>>(
          o16, IDD, 0, 0, wo16, DDIM, 0, 0, bo + (size_t)l * DDIM,
          xbuf, 0, 0, nullptr, 0, 0, 1.f, xbuf, nullptr, DDIM, 0, 0, IDD, 1);
    }

    // ---- LN2 ----
    ln_kernel<<<dim3((unsigned)R), blk256, 0, stream>>>(
        xbuf, ln2_g + l * DDIM, ln2_b + l * DDIM, hbuf, h16);

    // ---- pw1: y = h @ pw1_w^T + pw1_b ----
    {
      dim3 grid((2 * INNER_C) / 64, (unsigned)(R / 128), 1);
      gemm_wmma<false, true><<<grid, blk128, 0, stream>>>(
          h16, DDIM, 0, 0, p116, DDIM, 0, 0, pw1_b + (size_t)l * 2 * INNER_C,
          nullptr, 0, 0, nullptr, 0, 0, 1.f, yb, nullptr, 2 * INNER_C, 0, 0,
          DDIM, 1);
    }
    glu_kernel<<<dim3((unsigned)((R * INNER_C + 255) / 256)), blk256, 0, stream>>>(yb, y2b);
    dwconv_kernel<<<dim3((unsigned)((R * INNER_C + 255) / 256)), blk256, 0, stream>>>(
        y2b, dw_w + (size_t)l * INNER_C * KW, dw_b + (size_t)l * INNER_C, dw16);

    // ---- pw2: x = x + dw @ pw2_w^T + pw2_b ----
    {
      dim3 grid(DDIM / 64, (unsigned)(R / 128), 1);
      gemm_wmma<false, true><<<grid, blk128, 0, stream>>>(
          dw16, INNER_C, 0, 0, p216, INNER_C, 0, 0, pw2_b + (size_t)l * DDIM,
          xbuf, 0, 0, nullptr, 0, 0, 1.f, xbuf, nullptr, DDIM, 0, 0, INNER_C, 1);
    }
  }

  hipMemcpyAsync(d_out, xbuf, RD * sizeof(float), hipMemcpyDeviceToDevice, stream);
}
